// MicroConv_67568425500741
// MI455X (gfx1250) — matrix operand order repaired and checked by reading, not compiled, not run on hardware
//
#include <hip/hip_runtime.h>

#define DIN 128
#define KH 8
#define DOUT 16
#define NEG_SLOPE 0.2f

typedef __attribute__((ext_vector_type(16))) __bf16 v16bf;
typedef __attribute__((ext_vector_type(8)))  float  v8f;

__device__ __forceinline__ unsigned short f2bf(float f) {
  unsigned u = __float_as_uint(f);
  u += 0x7fffu + ((u >> 16) & 1u);        // round-to-nearest-even
  return (unsigned short)(u >> 16);
}

// order-preserving involution float-bits <-> sortable signed int
__device__ __forceinline__ int enc_order(int i) { return i ^ ((i >> 31) & 0x7fffffff); }

// ---------------------------------------------------------------------------
// Pre-pass: convert W (fp32, [k][n]) to bf16, transposed to [n][k], once.
// ---------------------------------------------------------------------------
__global__ void w_to_bf16(const float* __restrict__ Ws, const float* __restrict__ Wd,
                          unsigned short* __restrict__ Wbs, unsigned short* __restrict__ Wbd)
{
  int i = blockIdx.x * blockDim.x + threadIdx.x;
  if (i >= DIN * DIN) return;
  int k = i >> 7, n = i & 127;
  Wbs[n * DIN + k] = f2bf(Ws[i]);
  Wbd[n * DIN + k] = f2bf(Wd[i]);
}

// ---------------------------------------------------------------------------
// GEMM: h[row,:] = feat[row,:] @ W + b  (bf16 WMMA, f32 acc)
// Fused epilogue: elr[row,k] = sum_d h[row,k*16+d] * attn[k,off+d]
// block = 128 threads = 4 waves; each wave computes a 16x128 strip; tile = 64 rows
// W arrives pre-converted bf16 [n][k]; pulled to LDS via async-to-LDS (ASYNCcnt).
// ---------------------------------------------------------------------------
__global__ __launch_bounds__(128) void gat_gemm_wmma(
    const float* __restrict__ feat, const unsigned short* __restrict__ Wb,
    const float* __restrict__ bias, const float* __restrict__ attn, int att_off,
    float* __restrict__ h, float* __restrict__ elr, int n_rows)
{
  __shared__ unsigned short sW[128][136];  // sW[n][k], +8 pad breaks bank conflicts
  __shared__ unsigned short sF[64][136];   // feature tile sF[r][c]
  const int tid  = threadIdx.x;
  const int row0 = blockIdx.x * 64;

  // async copy bf16 W straight into LDS: 2048 x 16B chunks, padded row layout
  {
    unsigned lds_base = (unsigned)(unsigned long long)&sW[0][0];
    for (int i = tid; i < 2048; i += 128) {
      int n = i >> 4, c = i & 15;                       // chunk c covers k = c*8..c*8+7
      unsigned ldsa = lds_base + (unsigned)(n * 272 + c * 16);
      unsigned voff = (unsigned)(i * 16);
      asm volatile("global_load_async_to_lds_b128 %0, %1, %2"
                   :: "v"(ldsa), "v"(voff), "s"(Wb) : "memory");
    }
  }
  // stage feature tile (fp32 -> bf16) while the async W copy is in flight
  for (int i = tid; i < 64 * 128; i += 128) {
    int r = i >> 7, c = i & 127;
    float v = (row0 + r < n_rows) ? feat[(size_t)(row0 + r) * DIN + c] : 0.0f;
    sF[r][c] = f2bf(v);
  }
  asm volatile("s_wait_asynccnt 0x0" ::: "memory");
  __syncthreads();

  const int wave = tid >> 5;
  const int lane = tid & 31;
  const int half = lane >> 4;    // wave32: lanes 0-15 vs 16-31
  const int l16  = lane & 15;
  const int wr   = wave * 16;

  v8f acc[8] = {};
  union Frag { v16bf v; unsigned u[8]; };

  #pragma unroll
  for (int kk = 0; kk < 4; ++kk) {          // K chunks of 32
    Frag A;
    #pragma unroll
    for (int v = 0; v < 8; ++v) {
      // 16-bit A 16x32 layout: VGPR v<4 -> K = 2v (+8 hi half); v>=4 -> K = 16+2(v-4) (+8)
      int k0 = kk * 32 + half * 8 + ((v < 4) ? (v * 2) : (16 + (v - 4) * 2));
      A.u[v] = *(const unsigned*)&sF[wr + l16][k0];
    }
    #pragma unroll
    for (int n = 0; n < 8; ++n) {           // 8 column tiles of 16 (== 8 heads)
      Frag B;
      #pragma unroll
      for (int v = 0; v < 8; ++v) {
        // 16-bit B 32x16 layout: lane = col, VGPR v<4 -> K = 2v (+8 hi half); v>=4 -> K = 16+2(v-4) (+8)
        int k0 = kk * 32 + half * 8 + ((v < 4) ? (v * 2) : (16 + (v - 4) * 2));
        B.u[v] = *(const unsigned*)&sW[n * 16 + l16][k0];
      }
      acc[n] = __builtin_amdgcn_wmma_f32_16x16x32_bf16(
          false, A.v, false, B.v, (short)0, acc[n], false, false);
    }
  }

  // epilogue: store h (optional) + fused attention-logit reduction.
  // C/D layout: lane l16 = col-in-tile, VGPR r -> M = r (+8 for lanes 16-31).
  // head n == column tile n since DOUT == 16.
  #pragma unroll
  for (int n = 0; n < 8; ++n) {
    int col = n * DOUT + l16;
    float bv = bias[col];
    float aw = attn[n * (2 * DOUT) + att_off + l16];
    #pragma unroll
    for (int r = 0; r < 8; ++r) {
      int rowg = row0 + wr + r + half * 8;
      float hv = acc[n][r] + bv;
      if (h && rowg < n_rows) h[(size_t)rowg * DIN + col] = hv;
      float p = hv * aw;                    // reduce over the 16 lanes of this half
      p += __shfl_xor(p, 8, 16);
      p += __shfl_xor(p, 4, 16);
      p += __shfl_xor(p, 2, 16);
      p += __shfl_xor(p, 1, 16);
      if (l16 == 0 && rowg < n_rows) elr[rowg * KH + n] = p;
    }
  }
}

// init out=0, esum=0, emax = enc(-inf)
__global__ void gat_init(float* __restrict__ out, float* __restrict__ esum,
                         int* __restrict__ emax, int n_dst)
{
  int i = blockIdx.x * blockDim.x + threadIdx.x;
  if (i < n_dst * DIN) out[i] = 0.f;
  if (i < n_dst * KH) { esum[i] = 0.f; emax[i] = (int)0x807fffffu; }
}

// pass 1: e = leakyrelu(el[src]+er[dst]); store; segment-max via int atomicMax
__global__ void gat_edge1(const int* __restrict__ src, const int* __restrict__ dst,
                          const float* __restrict__ el, const float* __restrict__ er,
                          float* __restrict__ ebuf, int* __restrict__ emax, int E)
{
  int e = blockIdx.x * blockDim.x + threadIdx.x;
  if (e >= E) return;
  int s = src[e], d = dst[e];
  #pragma unroll
  for (int k = 0; k < KH; ++k) {
    float v = el[s * KH + k] + er[d * KH + k];
    v = (v >= 0.f) ? v : NEG_SLOPE * v;
    ebuf[(size_t)e * KH + k] = v;
    atomicMax(&emax[d * KH + k], enc_order(__float_as_int(v)));
  }
}

// decode emax, replicate reference isfinite guard (in-place: int buffer -> float)
__global__ void gat_nodemax(int* __restrict__ emax_bits, float* __restrict__ emaxf, int n)
{
  int i = blockIdx.x * blockDim.x + threadIdx.x;
  if (i >= n) return;
  float f = __int_as_float(enc_order(emax_bits[i]));
  if (!isfinite(f)) f = 0.f;
  emaxf[i] = f;
}

// pass 2: ee = exp(e - emax[dst]); store; segment-sum via float atomicAdd
__global__ void gat_edge2(const int* __restrict__ dst, float* __restrict__ ebuf,
                          const float* __restrict__ emaxf, float* __restrict__ esum, int E)
{
  int e = blockIdx.x * blockDim.x + threadIdx.x;
  if (e >= E) return;
  int d = dst[e];
  #pragma unroll
  for (int k = 0; k < KH; ++k) {
    float ee = __expf(ebuf[(size_t)e * KH + k] - emaxf[d * KH + k]);
    ebuf[(size_t)e * KH + k] = ee;
    atomicAdd(&esum[d * KH + k], ee);
  }
}

// pass 3: out[dst, c] += (ee/esum[dst]) * hs[src, c]; one lane per (edge, channel)
__global__ void gat_edge3(const int* __restrict__ src, const int* __restrict__ dst,
                          const float* __restrict__ ebuf, const float* __restrict__ esum,
                          const float* __restrict__ hs, float* __restrict__ out, int E)
{
  int gid = (int)(blockIdx.x * 256u + threadIdx.x);
  if (gid >= E * DIN) return;
  int e = gid >> 7, c = gid & 127;
  int s = src[e], d = dst[e];
  int k = c >> 4;
  float a = ebuf[(size_t)e * KH + k] / esum[d * KH + k];
  atomicAdd(&out[(size_t)d * DIN + c], a * hs[(size_t)s * DIN + c]);
}

// ---------------------------------------------------------------------------
extern "C" void kernel_launch(void* const* d_in, const int* in_sizes, int n_in,
                              void* d_out, int out_size, void* d_ws, size_t ws_size,
                              hipStream_t stream)
{
  const float* feat_src = (const float*)d_in[0];
  const float* feat_dst = (const float*)d_in[1];
  const float* W_src    = (const float*)d_in[2];
  const float* b_src    = (const float*)d_in[3];
  const float* W_dst    = (const float*)d_in[4];
  const float* b_dst    = (const float*)d_in[5];
  const float* attn     = (const float*)d_in[6];
  const int*   src_idx  = (const int*)d_in[7];
  const int*   dst_idx  = (const int*)d_in[8];

  const int n_src = in_sizes[0] / DIN;
  const int n_dst = in_sizes[1] / DIN;
  const int E     = in_sizes[7];
  float* out = (float*)d_out;

  char* ws = (char*)d_ws;
  size_t off = 0;
  auto take = [&](size_t bytes) -> char* {
    char* p = ws + off;
    off += (bytes + 255) & ~(size_t)255;
    return p;
  };
  float* hs   = (float*)take((size_t)n_src * DIN * sizeof(float));
  float* el   = (float*)take((size_t)n_src * KH * sizeof(float));
  float* er   = (float*)take((size_t)n_dst * KH * sizeof(float));
  int*   emax = (int*)  take((size_t)n_dst * KH * sizeof(int));
  float* esum = (float*)take((size_t)n_dst * KH * sizeof(float));
  float* ebuf = (float*)take((size_t)E * KH * sizeof(float));
  unsigned short* Wbs = (unsigned short*)take((size_t)DIN * DIN * sizeof(unsigned short));
  unsigned short* Wbd = (unsigned short*)take((size_t)DIN * DIN * sizeof(unsigned short));
  (void)ws_size; (void)n_in; (void)out_size;

  const int T = 256;
  w_to_bf16<<<dim3((DIN * DIN + T - 1) / T), dim3(T), 0, stream>>>(W_src, W_dst, Wbs, Wbd);

  // src projection: store hs + el.  dst projection: logits only (hd never stored).
  gat_gemm_wmma<<<dim3((n_src + 63) / 64), dim3(128), 0, stream>>>(
      feat_src, Wbs, b_src, attn, 0,    hs,      el, n_src);
  gat_gemm_wmma<<<dim3((n_dst + 63) / 64), dim3(128), 0, stream>>>(
      feat_dst, Wbd, b_dst, attn, DOUT, nullptr, er, n_dst);

  gat_init<<<dim3(((unsigned)n_dst * DIN + T - 1) / T), dim3(T), 0, stream>>>(out, esum, emax, n_dst);

  gat_edge1<<<dim3((E + T - 1) / T), dim3(T), 0, stream>>>(src_idx, dst_idx, el, er, ebuf, emax, E);
  gat_nodemax<<<dim3((n_dst * KH + T - 1) / T), dim3(T), 0, stream>>>(emax, (float*)emax, n_dst * KH);
  gat_edge2<<<dim3((E + T - 1) / T), dim3(T), 0, stream>>>(dst_idx, ebuf, (const float*)emax, esum, E);

  long long tot3 = (long long)E * DIN;
  gat_edge3<<<dim3((unsigned)((tot3 + 255) / 256)), dim3(256), 0, stream>>>(
      src_idx, dst_idx, ebuf, esum, hs, out, E);
}
